// PosoForMHALayer_13907104104568
// MI455X (gfx1250) — matrix operand order repaired and checked by reading, not compiled, not run on hardware
//
#include <hip/hip_runtime.h>

#define DEVINL __device__ __forceinline__

// Set to 0 to fall back to the synchronous LDS staging.
#ifndef USE_ASYNC_LDS
#define USE_ASYNC_LDS 1
#endif

typedef __attribute__((ext_vector_type(16))) __bf16 v16bf;
typedef __attribute__((ext_vector_type(8)))  float  v8f;
typedef __attribute__((ext_vector_type(4)))  float  v4f;
typedef __attribute__((ext_vector_type(4)))  unsigned int v4u;
typedef __attribute__((ext_vector_type(4)))  unsigned short v4us;
typedef unsigned short u16;

constexpr int B_ = 8, S_ = 2048, D_ = 512, H_ = 8, NV_ = 6, GIN_ = 64;
constexpr int HD_ = D_ / H_;            // 64
constexpr int GK_H_ = 128, GV_H_ = 32;
constexpr int BM = 64, BN = 64, BK = 32;

union FragAB { v16bf bf; v4u q[2]; };

DEVINL u16 f32_to_bf16(float f) {
  unsigned int x = __float_as_uint(f);
  x += 0x7FFFu + ((x >> 16) & 1u);       // round-to-nearest-even
  return (u16)(x >> 16);
}

DEVINL v8f wmma_bf16(const FragAB& a, const FragAB& b, v8f c) {
  return __builtin_amdgcn_wmma_f32_16x16x32_bf16(
      false, a.bf, false, b.bf, (short)0, c, false, false);
}

// --- CDNA5 async global->LDS copy (ASYNCcnt path, no VGPR round-trip) -------
DEVINL void async_copy_b128(void* lds_dst, const void* gsrc) {
#if USE_ASYNC_LDS
  // Low 32 bits of a generic shared pointer are the DS-space address.
  unsigned lds = (unsigned)(uintptr_t)lds_dst;
  unsigned long long ga = (unsigned long long)(uintptr_t)gsrc;
  asm volatile("global_load_async_to_lds_b128 %0, %1, off"
               :: "v"(lds), "v"(ga) : "memory");
#else
  *(v4u*)lds_dst = *(const v4u*)gsrc;
#endif
}

DEVINL void wait_async0() {
#if USE_ASYNC_LDS
#if __has_builtin(__builtin_amdgcn_s_wait_asynccnt)
  __builtin_amdgcn_s_wait_asynccnt(0);
#else
  asm volatile("s_wait_asynccnt 0x0" ::: "memory");
#endif
#endif
}

// ---------------------------------------------------------------------------
// Gate MLPs: gate_k[B,D], gate_v[B,NV], bmix[b,d] = sum_n gate_v[b,n]*bv[n,d]
// ---------------------------------------------------------------------------
__global__ __launch_bounds__(128)
void gates_kernel(const float* __restrict__ x,
                  const float* __restrict__ gkw1, const float* __restrict__ gkb1,
                  const float* __restrict__ gkw2, const float* __restrict__ gkb2,
                  const float* __restrict__ gvw1, const float* __restrict__ gvb1,
                  const float* __restrict__ gvw2, const float* __restrict__ gvb2,
                  const float* __restrict__ bv,
                  float* __restrict__ gateK, float* __restrict__ gateV,
                  float* __restrict__ bmix) {
  const int b = blockIdx.x, tid = threadIdx.x;
  __shared__ float xs[GIN_];
  __shared__ float hk[GK_H_];
  __shared__ float hv[GV_H_];
  __shared__ float gvs[NV_];
  if (tid < GIN_) xs[tid] = x[b * GIN_ + tid];
  __syncthreads();
  if (tid < GK_H_) {
    float s = gkb1[tid];
    for (int i = 0; i < GIN_; ++i) s += xs[i] * gkw1[i * GK_H_ + tid];
    hk[tid] = fmaxf(s, 0.f);
  }
  if (tid < GV_H_) {
    float s = gvb1[tid];
    for (int i = 0; i < GIN_; ++i) s += xs[i] * gvw1[i * GV_H_ + tid];
    hv[tid] = fmaxf(s, 0.f);
  }
  __syncthreads();
  for (int d = tid; d < D_; d += 128) {
    float s = gkb2[d];
    for (int j = 0; j < GK_H_; ++j) s += hk[j] * gkw2[j * D_ + d];
    gateK[b * D_ + d] = 2.f / (1.f + __expf(-s));
  }
  if (tid < NV_) {
    float s = gvb2[tid];
    for (int j = 0; j < GV_H_; ++j) s += hv[j] * gvw2[j * NV_ + tid];
    float g = 2.f / (1.f + __expf(-s));
    gvs[tid] = g;
    gateV[b * NV_ + tid] = g;
  }
  __syncthreads();
  for (int d = tid; d < D_; d += 128) {
    float s = 0.f;
    for (int n = 0; n < NV_; ++n) s += gvs[n] * bv[n * D_ + d];
    bmix[b * D_ + d] = s;
  }
}

// ---------------------------------------------------------------------------
// W[k][n] f32 -> Wt[n][k] bf16  (N-major so WMMA B-frags are contiguous in K)
// ---------------------------------------------------------------------------
__global__ __launch_bounds__(256)
void transpose_to_bf16(const float* __restrict__ W, u16* __restrict__ Wt) {
  __shared__ u16 tile[64][65];
  const int k0 = blockIdx.x * 64, n0 = blockIdx.y * 64, tid = threadIdx.x;
  #pragma unroll
  for (int e = 0; e < 16; ++e) {
    int idx = tid + e * 256;
    int kk = idx >> 6, nn = idx & 63;
    tile[nn][kk] = f32_to_bf16(W[(size_t)(k0 + kk) * D_ + n0 + nn]);
  }
  __syncthreads();
  #pragma unroll
  for (int e = 0; e < 16; ++e) {
    int idx = tid + e * 256;
    int nn = idx >> 6, kk = idx & 63;
    Wt[(size_t)(n0 + nn) * D_ + k0 + kk] = tile[nn][kk];
  }
}

// ---------------------------------------------------------------------------
// WmixT[b][n][k] = bf16( sum_n' gate_v[b,n'] * Wv[n'][k][n] )
// ---------------------------------------------------------------------------
__global__ __launch_bounds__(256)
void mix_wv_transpose(const float* __restrict__ Wv, const float* __restrict__ gateV,
                      u16* __restrict__ WmixT) {
  const int b = blockIdx.z, tid = threadIdx.x;
  __shared__ u16 tile[64][65];
  __shared__ float g[NV_];
  if (tid < NV_) g[tid] = gateV[b * NV_ + tid];
  __syncthreads();
  const int k0 = blockIdx.x * 64, n0 = blockIdx.y * 64;
  #pragma unroll
  for (int e = 0; e < 16; ++e) {
    int idx = tid + e * 256;
    int kk = idx >> 6, nn = idx & 63;
    size_t off = (size_t)(k0 + kk) * D_ + (n0 + nn);
    float s = 0.f;
    #pragma unroll
    for (int n = 0; n < NV_; ++n) s += g[n] * Wv[(size_t)n * D_ * D_ + off];
    tile[nn][kk] = f32_to_bf16(s);
  }
  __syncthreads();
  #pragma unroll
  for (int e = 0; e < 16; ++e) {
    int idx = tid + e * 256;
    int nn = idx >> 6, kk = idx & 63;
    WmixT[(size_t)b * D_ * D_ + (size_t)(n0 + nn) * D_ + k0 + kk] = tile[nn][kk];
  }
}

// ---------------------------------------------------------------------------
// Batched GEMM: C[b] = (A[b] @ Bt^T + bias) * colscale   (WMMA bf16, 64x64 tile)
//   A: f32 or bf16 [M,K] row-major; Bt: bf16 [N,K] (N-major); C: bf16 or f32.
//   B tile (and bf16 A tile) staged via async global->LDS DMA; the f32 A
//   convert overlaps with the in-flight async copies.
// ---------------------------------------------------------------------------
template <bool A_F32, bool OUT_F32>
__global__ __launch_bounds__(128)
void gemm_wmma(const void* __restrict__ Araw, const u16* __restrict__ Bt,
               const float* __restrict__ bias, int bias_stride,
               const float* __restrict__ colscale, int cs_stride, float uscale,
               void* __restrict__ Cout, int K, int Nld,
               long long strideA, long long strideB, long long strideC) {
  const int b = blockIdx.z;
  const int n0 = blockIdx.x * BN, m0 = blockIdx.y * BM;
  const int tid = threadIdx.x;
  const int wave = tid >> 5, lane = tid & 31, half = lane >> 4, lm = lane & 15;

  __shared__ u16 As[BM * BK];
  __shared__ u16 Bs[BN * BK];

  const u16* BtB = Bt + (size_t)b * strideB;

  v8f acc[4];
  #pragma unroll
  for (int t = 0; t < 4; ++t)
    #pragma unroll
    for (int e = 0; e < 8; ++e) acc[t][e] = 0.f;

  for (int k0 = 0; k0 < K; k0 += BK) {
    // Issue async B-tile copies first so they overlap the A staging below.
    #pragma unroll
    for (int it = 0; it < 2; ++it) {
      int g = tid + it * 128;
      int n = g >> 2, k8 = (g & 3) * 8;
      async_copy_b128(&Bs[n * BK + k8], BtB + (size_t)(n0 + n) * K + k0 + k8);
    }
    if (A_F32) {
      const float* A = (const float*)Araw + (size_t)b * strideA;
      #pragma unroll
      for (int it = 0; it < 4; ++it) {
        int g = tid + it * 128;                 // 512 float4-groups
        int m = g >> 3, k4 = (g & 7) * 4;
        v4f af = *(const v4f*)(A + (size_t)(m0 + m) * K + k0 + k4);
        v4us bfv;
        #pragma unroll
        for (int j = 0; j < 4; ++j) bfv[j] = f32_to_bf16(af[j]);
        *(v4us*)&As[m * BK + k4] = bfv;
      }
    } else {
      const u16* A = (const u16*)Araw + (size_t)b * strideA;
      #pragma unroll
      for (int it = 0; it < 2; ++it) {
        int g = tid + it * 128;                 // 256 16B-groups
        int m = g >> 2, k8 = (g & 3) * 8;
        async_copy_b128(&As[m * BK + k8], A + (size_t)(m0 + m) * K + k0 + k8);
      }
    }
    wait_async0();
    __syncthreads();

    FragAB a;                                    // A: row = wave*16+lm
    a.q[0] = *(const v4u*)&As[(wave * 16 + lm) * BK + half * 8];
    a.q[1] = *(const v4u*)&As[(wave * 16 + lm) * BK + 16 + half * 8];
    #pragma unroll
    for (int nt = 0; nt < 4; ++nt) {
      FragAB bf;                                 // B: col = nt*16+lm, K half*8.., 16+half*8..
      bf.q[0] = *(const v4u*)&Bs[(nt * 16 + lm) * BK + half * 8];
      bf.q[1] = *(const v4u*)&Bs[(nt * 16 + lm) * BK + 16 + half * 8];
      acc[nt] = wmma_bf16(a, bf, acc[nt]);
    }
    __syncthreads();
  }

  #pragma unroll
  for (int nt = 0; nt < 4; ++nt) {
    int n = n0 + nt * 16 + lm;
    float bsv = bias ? bias[(size_t)b * bias_stride + n] : 0.f;
    float sc = colscale ? colscale[(size_t)b * cs_stride + n] : uscale;
    #pragma unroll
    for (int r = 0; r < 8; ++r) {
      int m = m0 + wave * 16 + half * 8 + r;     // C layout: VGPR r -> row r / r+8
      float v = (acc[nt][r] + bsv) * sc;
      size_t idx = (size_t)b * strideC + (size_t)m * Nld + n;
      if (OUT_F32) ((float*)Cout)[idx] = v;
      else         ((u16*)Cout)[idx] = f32_to_bf16(v);
    }
  }
}

// ---------------------------------------------------------------------------
// Flash attention: per (b, h, 64 q-rows). Q LDS-resident (async DMA); 64-key
// steps: K tile via async global->LDS DMA overlapping the V VGPR-transpose;
// online softmax; P re-enters A-frag layout via per-wave LDS (same-wave
// in-order DS, no barrier). 1/sqrt(HD) already folded into qp.
// ---------------------------------------------------------------------------
__global__ __launch_bounds__(128)
void flash_attn(const u16* __restrict__ qp, const u16* __restrict__ kp,
                const u16* __restrict__ vp, const float* __restrict__ mask,
                const float* __restrict__ Cptr, u16* __restrict__ outp) {
  const int b = blockIdx.z, h = blockIdx.y, q0 = blockIdx.x * 64, d0 = h * HD_;
  const int tid = threadIdx.x;
  const int wave = tid >> 5, lane = tid & 31, half = lane >> 4, lm = lane & 15;

  __shared__ u16 Qs[64 * 64];
  __shared__ u16 Ks[64 * 64];
  __shared__ u16 Vt[64 * 64];                    // Vt[hd][kv]
  __shared__ u16 Ps[4][16 * 64];                 // per-wave P tile

  const float Cval = Cptr[0];

  const size_t baseQ = ((size_t)b * S_ + q0) * D_ + d0;
  #pragma unroll
  for (int it = 0; it < 4; ++it) {
    int g = tid + it * 128;
    int qr = g >> 3, d8 = (g & 7) * 8;
    async_copy_b128(&Qs[qr * 64 + d8], qp + baseQ + (size_t)qr * D_ + d8);
  }

  float mstate[8], lstate[8];
  v8f o[4];
  #pragma unroll
  for (int r = 0; r < 8; ++r) { mstate[r] = -1e30f; lstate[r] = 0.f; }
  #pragma unroll
  for (int t = 0; t < 4; ++t)
    #pragma unroll
    for (int e = 0; e < 8; ++e) o[t][e] = 0.f;

  for (int k0 = 0; k0 < S_; k0 += 64) {
    __syncthreads();                              // Ks/Vt safe to overwrite
    const size_t baseK = ((size_t)b * S_ + k0) * D_ + d0;
    // Async K-tile DMA overlaps the V transpose staging below.
    #pragma unroll
    for (int it = 0; it < 4; ++it) {
      int g = tid + it * 128;
      int kr = g >> 3, d8 = (g & 7) * 8;
      async_copy_b128(&Ks[kr * 64 + d8], kp + baseK + (size_t)kr * D_ + d8);
    }
    #pragma unroll
    for (int it = 0; it < 4; ++it) {
      int g = tid + it * 128;
      int kr = g >> 3, d8 = (g & 7) * 8;
      v4u vv = *(const v4u*)(vp + baseK + (size_t)kr * D_ + d8);
      const u16* pv = (const u16*)&vv;
      #pragma unroll
      for (int j = 0; j < 8; ++j) Vt[(d8 + j) * 64 + kr] = pv[j];
    }
    wait_async0();                                // covers Q DMA on first iter
    __syncthreads();

    // S = Q @ K^T : kp row-major [kv][d] is exactly the N-major B layout.
    v8f sacc[4];
    #pragma unroll
    for (int nt = 0; nt < 4; ++nt)
      #pragma unroll
      for (int e = 0; e < 8; ++e) sacc[nt][e] = 0.f;
    #pragma unroll
    for (int c = 0; c < 2; ++c) {
      FragAB a;
      int qrow = wave * 16 + lm;
      a.q[0] = *(const v4u*)&Qs[qrow * 64 + c * 32 + half * 8];
      a.q[1] = *(const v4u*)&Qs[qrow * 64 + c * 32 + 16 + half * 8];
      #pragma unroll
      for (int nt = 0; nt < 4; ++nt) {
        FragAB kb;
        int krow = nt * 16 + lm;
        kb.q[0] = *(const v4u*)&Ks[krow * 64 + c * 32 + half * 8];
        kb.q[1] = *(const v4u*)&Ks[krow * 64 + c * 32 + 16 + half * 8];
        sacc[nt] = wmma_bf16(a, kb, sacc[nt]);
      }
    }

    // mask + online softmax (rows live across 16-lane halves -> xor 1,2,4,8)
    const size_t mbase = ((size_t)b * S_ + q0) * S_ + k0;
    #pragma unroll
    for (int r = 0; r < 8; ++r) {
      int qloc = wave * 16 + half * 8 + r;
      float rmax = -1e30f;
      #pragma unroll
      for (int nt = 0; nt < 4; ++nt) {
        int kloc = nt * 16 + lm;
        float s = sacc[nt][r] + mask[mbase + (size_t)qloc * S_ + kloc] * (-1e9f);
        sacc[nt][r] = s;
        rmax = fmaxf(rmax, s);
      }
      rmax = fmaxf(rmax, __shfl_xor(rmax, 1, 32));
      rmax = fmaxf(rmax, __shfl_xor(rmax, 2, 32));
      rmax = fmaxf(rmax, __shfl_xor(rmax, 4, 32));
      rmax = fmaxf(rmax, __shfl_xor(rmax, 8, 32));
      float mnew = fmaxf(mstate[r], rmax);
      float corr = __expf(mstate[r] - mnew);
      mstate[r] = mnew;
      float psum = 0.f;
      #pragma unroll
      for (int nt = 0; nt < 4; ++nt) {
        float pe = __expf(sacc[nt][r] - mnew);
        psum += pe;
        Ps[wave][(half * 8 + r) * 64 + nt * 16 + lm] = f32_to_bf16(pe);
      }
      psum += __shfl_xor(psum, 1, 32);
      psum += __shfl_xor(psum, 2, 32);
      psum += __shfl_xor(psum, 4, 32);
      psum += __shfl_xor(psum, 8, 32);
      lstate[r] = lstate[r] * corr + psum;
      #pragma unroll
      for (int t = 0; t < 4; ++t) o[t][r] *= corr;
    }

    // O += P @ V  (Ps: same-wave LDS writes are in-order with these reads)
    #pragma unroll
    for (int c = 0; c < 2; ++c) {
      FragAB pa;
      pa.q[0] = *(const v4u*)&Ps[wave][lm * 64 + c * 32 + half * 8];
      pa.q[1] = *(const v4u*)&Ps[wave][lm * 64 + c * 32 + 16 + half * 8];
      #pragma unroll
      for (int t = 0; t < 4; ++t) {
        FragAB vb;
        vb.q[0] = *(const v4u*)&Vt[(t * 16 + lm) * 64 + c * 32 + half * 8];
        vb.q[1] = *(const v4u*)&Vt[(t * 16 + lm) * 64 + c * 32 + 16 + half * 8];
        o[t] = wmma_bf16(pa, vb, o[t]);
      }
    }
  }

  #pragma unroll
  for (int t = 0; t < 4; ++t)
    #pragma unroll
    for (int r = 0; r < 8; ++r) {
      int qloc = wave * 16 + half * 8 + r;
      float v = o[t][r] / lstate[r] * Cval;
      outp[((size_t)b * S_ + q0 + qloc) * D_ + d0 + t * 16 + lm] = f32_to_bf16(v);
    }
}

// ---------------------------------------------------------------------------
extern "C" void kernel_launch(void* const* d_in, const int* in_sizes, int n_in,
                              void* d_out, int out_size, void* d_ws, size_t ws_size,
                              hipStream_t stream) {
  (void)in_sizes; (void)n_in; (void)out_size; (void)ws_size;
  const float* x_gate = (const float*)d_in[0];
  const float* q    = (const float*)d_in[1];
  const float* k    = (const float*)d_in[2];
  const float* v    = (const float*)d_in[3];
  const float* mask = (const float*)d_in[4];
  const float* Wq   = (const float*)d_in[5];
  const float* bq   = (const float*)d_in[6];
  const float* Wk   = (const float*)d_in[7];
  const float* bk   = (const float*)d_in[8];
  const float* Wv   = (const float*)d_in[9];
  const float* bv   = (const float*)d_in[10];
  const float* Wo   = (const float*)d_in[11];
  const float* bo   = (const float*)d_in[12];
  const float* gkw1 = (const float*)d_in[13];
  const float* gkb1 = (const float*)d_in[14];
  const float* gkw2 = (const float*)d_in[15];
  const float* gkb2 = (const float*)d_in[16];
  const float* gvw1 = (const float*)d_in[17];
  const float* gvb1 = (const float*)d_in[18];
  const float* gvw2 = (const float*)d_in[19];
  const float* gvb2 = (const float*)d_in[20];
  const float* Cp   = (const float*)d_in[21];

  char* p = (char*)d_ws;
  auto alloc = [&](size_t bytes) {
    char* r = p;
    p += (bytes + 255) & ~(size_t)255;
    return r;
  };
  float* gateK = (float*)alloc((size_t)B_ * D_ * 4);
  float* gateV = (float*)alloc((size_t)B_ * NV_ * 4);
  float* bmix  = (float*)alloc((size_t)B_ * D_ * 4);
  u16* WqT   = (u16*)alloc((size_t)D_ * D_ * 2);
  u16* WkT   = (u16*)alloc((size_t)D_ * D_ * 2);
  u16* WoT   = (u16*)alloc((size_t)D_ * D_ * 2);
  u16* WmixT = (u16*)alloc((size_t)B_ * D_ * D_ * 2);
  u16* qp   = (u16*)alloc((size_t)B_ * S_ * D_ * 2);
  u16* kpb  = (u16*)alloc((size_t)B_ * S_ * D_ * 2);
  u16* vpb  = (u16*)alloc((size_t)B_ * S_ * D_ * 2);
  u16* attn = (u16*)alloc((size_t)B_ * S_ * D_ * 2);

  gates_kernel<<<B_, 128, 0, stream>>>(x_gate, gkw1, gkb1, gkw2, gkb2,
                                       gvw1, gvb1, gvw2, gvb2, bv,
                                       gateK, gateV, bmix);
  dim3 tgrid(D_ / 64, D_ / 64);
  transpose_to_bf16<<<tgrid, 256, 0, stream>>>(Wq, WqT);
  transpose_to_bf16<<<tgrid, 256, 0, stream>>>(Wk, WkT);
  transpose_to_bf16<<<tgrid, 256, 0, stream>>>(Wo, WoT);
  mix_wv_transpose<<<dim3(D_ / 64, D_ / 64, B_), 256, 0, stream>>>(Wv, gateV, WmixT);

  dim3 ggrid(D_ / BN, S_ / BM, B_);
  const long long sA = (long long)S_ * D_, sC = (long long)S_ * D_;
  // qp = (q@Wq + bq) * 0.125  (1/sqrt(HD) folded into Q)
  gemm_wmma<true, false><<<ggrid, 128, 0, stream>>>(
      q, WqT, bq, 0, nullptr, 0, 0.125f, qp, D_, D_, sA, 0, sC);
  // kp = (k@Wk + bk) * gate_k[b,:]
  gemm_wmma<true, false><<<ggrid, 128, 0, stream>>>(
      k, WkT, bk, 0, gateK, D_, 1.f, kpb, D_, D_, sA, 0, sC);
  // vp = v @ Wmix[b] + bmix[b]
  gemm_wmma<true, false><<<ggrid, 128, 0, stream>>>(
      v, WmixT, bmix, D_, nullptr, 0, 1.f, vpb, D_, D_, sA, (long long)D_ * D_, sC);

  flash_attn<<<dim3(S_ / 64, H_, B_), 128, 0, stream>>>(qp, kpb, vpb, mask, Cp, attn);

  // out = attn @ Wo + bo  (f32 output)
  gemm_wmma<false, true><<<ggrid, 128, 0, stream>>>(
      attn, WoT, bo, 0, nullptr, 0, 1.f, (float*)d_out, D_, D_, sA, 0, sC);
}
